// SubjectLayers_72215580115411
// MI455X (gfx1250) — matrix operand order repaired and checked by reading, compiled, not run on hardware
//
#include <hip/hip_runtime.h>

// Problem constants (from reference)
#define B_    64
#define T_    2048
#define CIN   256
#define COUT  256

// Tiling
#define BM    128
#define BN    128
#define BK    32
#define NP    16    // k-pairs per K tile (BK/2)
#define PS    20    // padded row stride (dwords): 16B-aligned rows, conflict-free frag reads

typedef __attribute__((ext_vector_type(16))) __bf16 v16bf;
typedef __attribute__((ext_vector_type(8)))  float  v8f;

union fragu { uint4 q[2]; v16bf b; };

// Pack two fp32 values (consecutive K) into fragment-ready dwords:
//   hp = { bf16_hi(b) [31:16] | bf16_hi(a) [15:0] }   (truncation split)
//   lp = { bf16_lo(b) [31:16] | bf16_lo(a) [15:0] }   lo = bf16(x - hi)
__device__ __forceinline__ void pack_pair(float a, float b, unsigned& hp, unsigned& lp) {
    unsigned ba = __float_as_uint(a);
    unsigned bb = __float_as_uint(b);
    unsigned ha = ba & 0xFFFF0000u;
    unsigned hb = bb & 0xFFFF0000u;
    float ra = a - __uint_as_float(ha);
    float rb = b - __uint_as_float(hb);
    hp = hb | (ha >> 16);
    lp = (__float_as_uint(rb) & 0xFFFF0000u) | (__float_as_uint(ra) >> 16);
}

__global__ __launch_bounds__(256)
void subject_linear_bf16x3(const float* __restrict__ x,
                           const int*   __restrict__ subject_id,
                           const float* __restrict__ weights,
                           const float* __restrict__ bias,
                           float* __restrict__ out) {
    // Pair-packed bf16 planes, fragment-ready layouts:
    //   xhs/xls: [m][p]  (A: row-major over k-pairs)
    //   whs/wls: [n][p]  (B: transposed -> contiguous per output column)
    __shared__ __align__(16) unsigned xhs[BM * PS];
    __shared__ __align__(16) unsigned xls[BM * PS];
    __shared__ __align__(16) unsigned whs[BN * PS];
    __shared__ __align__(16) unsigned wls[BN * PS];

    const int tid  = threadIdx.x;
    const int lane = tid & 31;
    const int w    = tid >> 5;
    const int wm   = w & 3;            // 4 waves along M
    const int wn   = w >> 2;           // 2 waves along N
    const int lm   = lane & 15;
    const int lh   = lane >> 4;

    const int mt = blockIdx.x;         // 0..15  (M tile)
    const int nt = blockIdx.y;         // 0..1   (N tile)
    const int b  = blockIdx.z;         // 0..63  (batch)
    const int s  = subject_id[b];

    const float* xg = x + ((size_t)b * T_ + (size_t)mt * BM) * CIN;
    const float* wg = weights + (size_t)s * CIN * COUT + (size_t)nt * BN;

    v8f acc[2][4] = {};

    for (int k0 = 0; k0 < CIN; k0 += BK) {
        // ---- stage x tile: 128 rows x 32 k -> pair-packed hi/lo planes ----
        #pragma unroll
        for (int it = 0; it < 4; ++it) {
            int idx = it * 256 + tid;          // 1024 float4 loads
            int m   = idx >> 3;                // row 0..127
            int cq  = idx & 7;                 // k-quad; k = k0 + 4*cq
            const float4 v = *(const float4*)(xg + (size_t)m * CIN + k0 + 4 * cq);
            unsigned h0, l0, h1, l1;
            pack_pair(v.x, v.y, h0, l0);       // pair p = 2*cq
            pack_pair(v.z, v.w, h1, l1);       // pair p = 2*cq + 1
            *(uint2*)&xhs[m * PS + 2 * cq] = make_uint2(h0, h1);
            *(uint2*)&xls[m * PS + 2 * cq] = make_uint2(l0, l1);
        }
        // ---- stage W tile transposed: 32 k x 128 n -> [n][p] hi/lo planes ----
        #pragma unroll
        for (int it = 0; it < 2; ++it) {
            int idx = it * 256 + tid;          // 512 units: (k-pair, n-quad)
            int p   = idx >> 5;                // pair 0..15  (k rows 2p, 2p+1)
            int nq  = idx & 31;                // n = 4*nq
            const float* g0 = wg + (size_t)(k0 + 2 * p) * COUT + 4 * nq;
            const float4 v0 = *(const float4*)(g0);
            const float4 v1 = *(const float4*)(g0 + COUT);
            int nb = 4 * nq;
            unsigned hp, lp;
            pack_pair(v0.x, v1.x, hp, lp); whs[(nb + 0) * PS + p] = hp; wls[(nb + 0) * PS + p] = lp;
            pack_pair(v0.y, v1.y, hp, lp); whs[(nb + 1) * PS + p] = hp; wls[(nb + 1) * PS + p] = lp;
            pack_pair(v0.z, v1.z, hp, lp); whs[(nb + 2) * PS + p] = hp; wls[(nb + 2) * PS + p] = lp;
            pack_pair(v0.w, v1.w, hp, lp); whs[(nb + 3) * PS + p] = hp; wls[(nb + 3) * PS + p] = lp;
        }
        // warm GL2 with next K tile while we compute this one
        if (k0 + BK < CIN) {
            __builtin_prefetch(xg + (size_t)(tid >> 3) * CIN + k0 + BK, 0, 1);
            __builtin_prefetch(wg + (size_t)(k0 + BK + (tid >> 5)) * COUT, 0, 1);
        }
        __syncthreads();

        // ---- B fragments: pure ds_load_b128, no repacking ----
        fragu bhi[4], blo[4];
        #pragma unroll
        for (int ni = 0; ni < 4; ++ni) {
            const unsigned* bh = &whs[(wn * 64 + ni * 16 + lm) * PS + lh * 4];
            const unsigned* bl = &wls[(wn * 64 + ni * 16 + lm) * PS + lh * 4];
            bhi[ni].q[0] = *(const uint4*)(bh);
            bhi[ni].q[1] = *(const uint4*)(bh + 8);
            blo[ni].q[0] = *(const uint4*)(bl);
            blo[ni].q[1] = *(const uint4*)(bl + 8);
        }
        // ---- A fragments + bf16x3 accumulation ----
        #pragma unroll
        for (int mi = 0; mi < 2; ++mi) {
            const unsigned* rh = &xhs[(wm * 32 + mi * 16 + lm) * PS];
            const unsigned* rl = &xls[(wm * 32 + mi * 16 + lm) * PS];
            fragu ahi, alo;
            // A 16x32 layout: VGPRs 0-3 <- pairs lh*4+0..3, VGPRs 4-7 <- pairs 8+lh*4+0..3
            ahi.q[0] = *(const uint4*)(rh + lh * 4);
            ahi.q[1] = *(const uint4*)(rh + 8 + lh * 4);
            alo.q[0] = *(const uint4*)(rl + lh * 4);
            alo.q[1] = *(const uint4*)(rl + 8 + lh * 4);
            #pragma unroll
            for (int ni = 0; ni < 4; ++ni) {
                acc[mi][ni] = __builtin_amdgcn_wmma_f32_16x16x32_bf16(
                    false, ahi.b, false, bhi[ni].b, (short)0, acc[mi][ni], false, false);
                acc[mi][ni] = __builtin_amdgcn_wmma_f32_16x16x32_bf16(
                    false, ahi.b, false, blo[ni].b, (short)0, acc[mi][ni], false, false);
                acc[mi][ni] = __builtin_amdgcn_wmma_f32_16x16x32_bf16(
                    false, alo.b, false, bhi[ni].b, (short)0, acc[mi][ni], false, false);
            }
        }
        __syncthreads();
    }

    // ---- epilogue: add gathered bias, store fp32 ----
    const float* bb = bias + (size_t)s * COUT;
    float* og = out + (size_t)b * T_ * COUT;
    #pragma unroll
    for (int mi = 0; mi < 2; ++mi) {
        int t0 = mt * BM + wm * 32 + mi * 16 + lh * 8;   // C/D layout: vgpr r -> M = r + 8*lh
        #pragma unroll
        for (int ni = 0; ni < 4; ++ni) {
            int n = nt * BN + wn * 64 + ni * 16 + lm;
            float bv = bb[n];
            #pragma unroll
            for (int r = 0; r < 8; ++r) {
                og[(size_t)(t0 + r) * COUT + n] = acc[mi][ni][r] + bv;
            }
        }
    }
}

extern "C" void kernel_launch(void* const* d_in, const int* in_sizes, int n_in,
                              void* d_out, int out_size, void* d_ws, size_t ws_size,
                              hipStream_t stream) {
    (void)in_sizes; (void)n_in; (void)out_size; (void)d_ws; (void)ws_size;
    const float* x   = (const float*)d_in[0];
    const int*   sid = (const int*)d_in[1];
    const float* wt  = (const float*)d_in[2];
    const float* bs  = (const float*)d_in[3];
    float* out = (float*)d_out;

    dim3 grid(T_ / BM, COUT / BN, B_);   // 16 x 2 x 64 = 2048 workgroups
    dim3 block(256);                     // 8 wave32s
    hipLaunchKernelGGL(subject_linear_bf16x3, grid, block, 0, stream,
                       x, sid, wt, bs, out);
}
